// ExperimentalLoss_23742579212660
// MI455X (gfx1250) — compile-verified, matrix operand order checked by reading, not run on hardware
//
#include <hip/hip_runtime.h>
#include <hip/hip_bf16.h>

// CDNA5 / gfx1250: wave32; WMMA 16x16x32 f16->f32 as the reduction engine;
// async global->LDS (ASYNCcnt) for tile streaming when the toolchain exposes it.

typedef __attribute__((ext_vector_type(16))) _Float16 v16h;
typedef __attribute__((ext_vector_type(8)))  float    v8f;

#define N_DIM 4096
#define TW 64            // tile width
#define TH 64            // tile height
#define LR 68            // loaded region (tile + 2-halo each side)
#define UR 66            // blurred region (tile + 1-halo each side)
#define DS_W 72          // padded LDS row stride (floats)
#define NBLK (64 * 64)   // 4096 tiles

#define AS1 __attribute__((address_space(1)))
#define AS3 __attribute__((address_space(3)))

#if defined(__has_builtin)
#  if __has_builtin(__builtin_amdgcn_global_load_async_to_lds_b32)
#    define USE_ASYNC_LDS 1
#  endif
#endif
#ifndef USE_ASYNC_LDS
#  define USE_ASYNC_LDS 0
#endif

__device__ __forceinline__ void wait_async0() {
#if defined(__has_builtin) && __has_builtin(__builtin_amdgcn_s_wait_asynccnt)
    __builtin_amdgcn_s_wait_asynccnt(0);
#else
    asm volatile("s_wait_asynccnt 0" ::: "memory");
#endif
}

// BORDER_REFLECT_101 index mirroring (jnp.pad mode='reflect')
__device__ __forceinline__ int mirror_idx(int x, int n) {
    x = (x < 0) ? -x : x;
    x = (x >= n) ? (2 * n - 2 - x) : x;
    return x;
}

__global__ __launch_bounds__(256) void loss_tile_kernel(
    const float* __restrict__ pred, const float* __restrict__ target,
    float* __restrict__ ws)
{
    // s_t: raw target tile, overwritten in place with diff = target - pred.
    // sp : raw pred tile, later reused (aliased) for the blurred field u.
    __shared__ float s_t[LR][DS_W];
    __shared__ union {
        float p[LR][DS_W];
        float u[UR][DS_W];
    } sp;
    __shared__ unsigned char valid_s[TH][TW];  // target > 0 mask (tile interior)
    __shared__ float red[3][256];

    const int tid = threadIdx.x;
    const int c0 = blockIdx.x * TW;
    const int r0 = blockIdx.y * TH;

    // ---- Phase 1a: stream raw tiles (+2 halo, mirrored borders) into LDS ----
    for (int idx = tid; idx < LR * LR; idx += 256) {
        const int lr = idx / LR, lc = idx % LR;
        const int gr = mirror_idx(r0 - 2 + lr, N_DIM);
        const int gc = mirror_idx(c0 - 2 + lc, N_DIM);
        const size_t off = (size_t)gr * N_DIM + gc;
#if USE_ASYNC_LDS
        // memory -> LDS directly, tracked by ASYNCcnt (no VGPR staging)
        __builtin_amdgcn_global_load_async_to_lds_b32(
            (AS1 int*)(target + off), (AS3 int*)&s_t[lr][lc], 0, 0);
        __builtin_amdgcn_global_load_async_to_lds_b32(
            (AS1 int*)(pred + off),   (AS3 int*)&sp.p[lr][lc], 0, 0);
#else
        const int nidx = idx + 256;               // prefetch next chunk
        if (nidx < LR * LR) {
            const int nlr = nidx / LR, nlc = nidx % LR;
            const int ngr = mirror_idx(r0 - 2 + nlr, N_DIM);
            const int ngc = mirror_idx(c0 - 2 + nlc, N_DIM);
            const size_t noff = (size_t)ngr * N_DIM + ngc;
            __builtin_prefetch(&pred[noff], 0, 1);
            __builtin_prefetch(&target[noff], 0, 1);
        }
        s_t[lr][lc]  = target[off];
        sp.p[lr][lc] = pred[off];
#endif
    }
#if USE_ASYNC_LDS
    wait_async0();
#endif
    __syncthreads();

    // ---- Phase 1b: valid mask, then diff in place over the target tile ----
    for (int idx = tid; idx < LR * LR; idx += 256) {
        const int lr = idx / LR, lc = idx % LR;
        const float tv = s_t[lr][lc];
        const float pv = sp.p[lr][lc];
        if (lr >= 2 && lr < 2 + TH && lc >= 2 && lc < 2 + TW)
            valid_s[lr - 2][lc - 2] = (tv > 0.0f) ? 1 : 0;
        s_t[lr][lc] = tv - pv;
    }
    __syncthreads();

    // ---- Phase 2: 3x3 gaussian ([1,2,1]/4 outer) over tile + 1 halo ----
    // writes sp.u, which aliases sp.p (raw pred fully consumed in phase 1b)
    for (int idx = tid; idx < UR * UR; idx += 256) {
        const int ur = idx / UR, uc = idx % UR;
        const float s =
            0.0625f * (s_t[ur][uc]     + s_t[ur][uc + 2] +
                       s_t[ur + 2][uc] + s_t[ur + 2][uc + 2]) +
            0.125f  * (s_t[ur][uc + 1] + s_t[ur + 1][uc] +
                       s_t[ur + 1][uc + 2] + s_t[ur + 2][uc + 1]) +
            0.25f   *  s_t[ur + 1][uc + 1];
        sp.u[ur][uc] = s;
    }
    __syncthreads();

    // ---- Phase 3: Sobel(dx=1,dy=1) corners + masked MSE terms, 16 px/thread ----
    const int c  = tid & (TW - 1);        // column within tile
    const int rb = (tid >> 6) * 16;       // 16-row strip
    const int C  = c0 + c;

    v16h ga, sa, va;                      // G, masked diff^2, valid flag (f16)
    const int b0 = mirror_idx(C - 1, N_DIM) - (c0 - 1);
    const int b1 = mirror_idx(C + 1, N_DIM) - (c0 - 1);
#pragma unroll
    for (int i = 0; i < 16; ++i) {
        const int r = rb + i;
        const int R = r0 + r;
        const int a0 = mirror_idx(R - 1, N_DIM) - (r0 - 1);
        const int a1 = mirror_idx(R + 1, N_DIM) - (r0 - 1);
        // sk = outer([-1,0,1],[-1,0,1]): only the 4 corners are nonzero
        const float G = sp.u[a0][b0] - sp.u[a0][b1] - sp.u[a1][b0] + sp.u[a1][b1];
        const float dv = s_t[r + 2][c + 2];            // diff (in place)
        const float vf = (float)valid_s[r][c];
        const float md = vf * dv;
        ga[i] = (_Float16)G;
        sa[i] = (_Float16)(md * md);
        va[i] = (_Float16)vf;
    }

    // ---- WMMA reduction: D = A x ones + 0 ; f32 accumulation in matrix core ----
    v16h onesv;
#pragma unroll
    for (int i = 0; i < 16; ++i) onesv[i] = (_Float16)1.0f;

    v8f z0 = {}, z1 = {}, z2 = {};
    v8f Dg = __builtin_amdgcn_wmma_f32_16x16x32_f16(false, ga, false, onesv,
                                                    (short)0, z0, false, false);
    v8f Ds = __builtin_amdgcn_wmma_f32_16x16x32_f16(false, sa, false, onesv,
                                                    (short)0, z1, false, false);
    v8f Dc = __builtin_amdgcn_wmma_f32_16x16x32_f16(false, va, false, onesv,
                                                    (short)0, z2, false, false);

    // each lane holds 8 of the 256 D entries; all-ones B => sum(D) = 16 * sum(A)
    float ps = 0.0f, pc = 0.0f, pg = 0.0f;
#pragma unroll
    for (int i = 0; i < 8; ++i) { ps += Ds[i]; pc += Dc[i]; pg += Dg[i]; }

    red[0][tid] = ps; red[1][tid] = pc; red[2][tid] = pg;
    __syncthreads();
    for (int s = 128; s > 0; s >>= 1) {
        if (tid < s) {
            red[0][tid] += red[0][tid + s];
            red[1][tid] += red[1][tid + s];
            red[2][tid] += red[2][tid + s];
        }
        __syncthreads();
    }
    if (tid == 0) {
        const int b = blockIdx.y * 64 + blockIdx.x;
        ws[b]            = red[0][0] * 0.0625f;  // sum_sq  (undo x16)
        ws[NBLK + b]     = red[1][0] * 0.0625f;  // count   (exact: /16 of 16*int)
        ws[2 * NBLK + b] = red[2][0] * 0.0625f;  // sum_G
    }
}

// Deterministic fixed-order final reduction (no float atomics anywhere).
__global__ __launch_bounds__(256) void finalize_kernel(
    const float* __restrict__ ws, float* __restrict__ out)
{
    __shared__ float red[3][256];
    const int tid = threadIdx.x;
    float s0 = 0.0f, s1 = 0.0f, s2 = 0.0f;
    for (int i = tid; i < NBLK; i += 256) {
        s0 += ws[i];
        s1 += ws[NBLK + i];
        s2 += ws[2 * NBLK + i];
    }
    red[0][tid] = s0; red[1][tid] = s1; red[2][tid] = s2;
    __syncthreads();
    for (int s = 128; s > 0; s >>= 1) {
        if (tid < s) {
            red[0][tid] += red[0][tid + s];
            red[1][tid] += red[1][tid + s];
            red[2][tid] += red[2][tid + s];
        }
        __syncthreads();
    }
    if (tid == 0) {
        const float sum_sq = red[0][0];
        const float cnt    = red[1][0];
        const float sum_g  = red[2][0];
        const float mse  = sum_sq / fmaxf(cnt, 1.0f);
        const float invN = 1.0f / ((float)N_DIM * (float)N_DIM);
        out[0] = 0.2f * (sum_g * invN) + 0.8f * mse;
    }
}

extern "C" void kernel_launch(void* const* d_in, const int* in_sizes, int n_in,
                              void* d_out, int out_size, void* d_ws, size_t ws_size,
                              hipStream_t stream) {
    const float* pred   = (const float*)d_in[0];
    const float* target = (const float*)d_in[1];
    float* ws  = (float*)d_ws;   // uses 3 * 4096 floats = 48 KB
    float* out = (float*)d_out;  // single f32 scalar

    dim3 grid(N_DIM / TW, N_DIM / TH);   // 64 x 64 tiles
    loss_tile_kernel<<<grid, 256, 0, stream>>>(pred, target, ws);
    finalize_kernel<<<1, 256, 0, stream>>>(ws, out);
}
